// ScaledDotProductAttention_60979945669130
// MI455X (gfx1250) — matrix operand order repaired and checked by reading, not compiled, not run on hardware
//
#include <hip/hip_runtime.h>

// CDNA5 / gfx1250 attention: out = softmax(QK^T/sqrt(D)) V, plus full weights.
// B=16, S=2048, D=128, fp32 in/out; bf16 WMMA (f32 accumulate) internally.
// Phase-3 V staging uses GLOBAL_LOAD_ASYNC_TO_LDS_B128 (ASYNCcnt-tracked direct
// LDS writes); falls back to VGPR staging if the builtin is unavailable.

typedef __attribute__((ext_vector_type(16))) __bf16 v16bf;
typedef __attribute__((ext_vector_type(8)))  float  v8f;
typedef __attribute__((ext_vector_type(4)))  int    v4i;

#define BATCH 16
#define SEQ   2048
#define DIM   128
#define SPAD  2052           // padded f16/bf16 score row stride (elements)
#define VPAD  132            // padded V-tile row stride (floats)

#if defined(__has_builtin)
#if __has_builtin(__builtin_amdgcn_global_load_async_to_lds_b128) && \
    __has_builtin(__builtin_amdgcn_s_wait_asynccnt)
#define HAVE_ASYNC_LDS 1
typedef __attribute__((address_space(1))) v4i* g_v4i_p;
typedef __attribute__((address_space(3))) v4i* l_v4i_p;
#endif
#endif

__device__ __forceinline__ unsigned short f16bits(float v) {
    return __builtin_bit_cast(unsigned short, (_Float16)v);
}
__device__ __forceinline__ float f16val(unsigned short u) {
    return (float)__builtin_bit_cast(_Float16, u);
}

// A-matrix frag (16x32 bf16, contiguous K order) from one fp32 row:
// element e in [0,16): K = kd + half*16 + e
__device__ __forceinline__ v16bf load_a_frag(const float* __restrict__ row, int kd, int half) {
    const float4* p0 = (const float4*)(row + kd + half * 16);
    const float4* p1 = (const float4*)(row + kd + half * 16 + 8);
    float4 a0 = p0[0], a1 = p0[1];
    float4 b0 = p1[0], b1 = p1[1];
    v16bf f;
    f[0]=(__bf16)a0.x; f[1]=(__bf16)a0.y; f[2]=(__bf16)a0.z; f[3]=(__bf16)a0.w;
    f[4]=(__bf16)a1.x; f[5]=(__bf16)a1.y; f[6]=(__bf16)a1.z; f[7]=(__bf16)a1.w;
    f[8]=(__bf16)b0.x; f[9]=(__bf16)b0.y; f[10]=(__bf16)b0.z; f[11]=(__bf16)b0.w;
    f[12]=(__bf16)b1.x; f[13]=(__bf16)b1.y; f[14]=(__bf16)b1.z; f[15]=(__bf16)b1.w;
    return f;
}

// B-matrix frag (32x16 bf16): element e -> K = kd + half*16 + e (16 consecutive floats)
__device__ __forceinline__ v16bf load_b_frag(const float* __restrict__ row, int kd, int half) {
    const float4* p = (const float4*)(row + kd + half * 16);
    float4 x0 = p[0], x1 = p[1], x2 = p[2], x3 = p[3];
    v16bf f;
    f[0]=(__bf16)x0.x; f[1]=(__bf16)x0.y; f[2]=(__bf16)x0.z; f[3]=(__bf16)x0.w;
    f[4]=(__bf16)x1.x; f[5]=(__bf16)x1.y; f[6]=(__bf16)x1.z; f[7]=(__bf16)x1.w;
    f[8]=(__bf16)x2.x; f[9]=(__bf16)x2.y; f[10]=(__bf16)x2.z; f[11]=(__bf16)x2.w;
    f[12]=(__bf16)x3.x; f[13]=(__bf16)x3.y; f[14]=(__bf16)x3.z; f[15]=(__bf16)x3.w;
    return f;
}

__global__ __launch_bounds__(256)
void attn_kernel(const float* __restrict__ Q, const float* __restrict__ K,
                 const float* __restrict__ V, float* __restrict__ out,
                 float* __restrict__ wts) {
    const int b    = blockIdx.y;
    const int q0   = blockIdx.x * 16;
    const int tid  = threadIdx.x;
    const int wave = tid >> 5;
    const int lane = tid & 31;
    const int half = lane >> 4;
    const int ln   = lane & 15;

    const float* Qb = Q + ((size_t)b * SEQ + q0) * DIM;
    const float* Kb = K + (size_t)b * SEQ * DIM;
    const float* Vb = V + (size_t)b * SEQ * DIM;
    float* outp = out + ((size_t)b * SEQ + q0) * DIM;
    float* wrow = wts + ((size_t)b * SEQ + q0) * (size_t)SEQ;

    // f16 scores -> overwritten in place with bf16 probs (16 rows x SEQ, padded)
    __shared__ __attribute__((aligned(16))) unsigned short sc[16 * SPAD];
    // double-buffered V tile: 32 rows x DIM floats, padded
    __shared__ __attribute__((aligned(16))) float vbuf[2][32 * VPAD];

    const float scale = 0.08838834764831845f;   // 1/sqrt(128)
    const float log2e = 1.4426950408889634f;

    // ---------------- Phase 1: scores = Q K^T * scale (bf16 WMMA, f32 acc) ----
    v16bf qf[4];
    {
        const float* qrow = Qb + (size_t)ln * DIM;
        #pragma unroll
        for (int s = 0; s < 4; ++s) qf[s] = load_a_frag(qrow, 32 * s, half);
    }

    for (int kt = wave; kt < SEQ / 16; kt += 8) {
        const int key0 = kt * 16;
        const float* krow = Kb + (size_t)(key0 + ln) * DIM;
        if (kt + 8 < SEQ / 16)
            __builtin_prefetch(Kb + (size_t)((kt + 8) * 16 + ln) * DIM, 0, 0);
        v8f acc = {};
        #pragma unroll
        for (int s = 0; s < 4; ++s) {
            v16bf kf = load_b_frag(krow, 32 * s, half);
            acc = __builtin_amdgcn_wmma_f32_16x16x32_bf16(
                false, qf[s], false, kf, (short)0, acc, false, false);
        }
        // C layout: element r -> row M = r + 8*half, col N = ln
        #pragma unroll
        for (int r = 0; r < 8; ++r) {
            const int m = r + 8 * half;
            sc[m * SPAD + key0 + ln] = f16bits(acc[r] * scale);
        }
    }
    __syncthreads();

    // ---------------- Phase 2: softmax rows; write weights; probs -> bf16 LDS -
    #pragma unroll
    for (int rr = 0; rr < 2; ++rr) {
        const int r = wave * 2 + rr;
        unsigned short* srow = sc + r * SPAD;
        float m = -INFINITY;
        for (int i = lane; i < SEQ; i += 32) m = fmaxf(m, f16val(srow[i]));
        #pragma unroll
        for (int off = 16; off; off >>= 1) m = fmaxf(m, __shfl_xor(m, off, 32));
        float l = 0.0f;
        for (int i = lane; i < SEQ; i += 32)
            l += exp2f((f16val(srow[i]) - m) * log2e);
        #pragma unroll
        for (int off = 16; off; off >>= 1) l += __shfl_xor(l, off, 32);
        const float inv = 1.0f / l;
        float* wdst = wrow + (size_t)r * SEQ;
        for (int i = lane; i < SEQ; i += 32) {
            const float p = exp2f((f16val(srow[i]) - m) * log2e) * inv;
            wdst[i] = p;                                             // fp32 weights out
            srow[i] = __builtin_bit_cast(unsigned short, (__bf16)p); // bf16 probs
        }
    }
    __syncthreads();

    // ---------------- Phase 3: out = P V  (LDS-staged V, double buffered) -----
    const int d = wave * 16 + ln;   // each wave owns a 16-wide slice of D
    v8f oacc = {};

    auto stage = [&](int step) {
        float* dst = vbuf[step & 1];
        const float* src = Vb + (size_t)step * 32 * DIM;  // 32 rows contiguous
        #pragma unroll
        for (int i = 0; i < 4; ++i) {
            const int f4  = i * 256 + tid;
            const int fl  = f4 * 4;
            const int row = fl >> 7;       // /DIM
            const int col = fl & (DIM - 1);
#if defined(HAVE_ASYNC_LDS)
            // ASYNCcnt-tracked direct global->LDS b128 copy (no VGPR round trip)
            __builtin_amdgcn_global_load_async_to_lds_b128(
                (g_v4i_p)(src + fl),
                (l_v4i_p)(dst + row * VPAD + col),
                0, 0);
#else
            *(float4*)(dst + row * VPAD + col) = *(const float4*)(src + fl);
#endif
        }
    };

    stage(0);
#if defined(HAVE_ASYNC_LDS)
    __builtin_amdgcn_s_wait_asynccnt(0);
#endif
    __syncthreads();
    for (int step = 0; step < SEQ / 32; ++step) {
        if (step + 1 < SEQ / 32) stage(step + 1);
        const float* vb = vbuf[step & 1];
        const int kb = step * 32;
        v16bf af, bfr;
        #pragma unroll
        for (int e = 0; e < 8; ++e) {
            const int ka = half * 16 + e;       // contiguous K index (first 8)
            af[e]     = __builtin_bit_cast(__bf16, sc[ln * SPAD + kb + ka]);
            af[8 + e] = __builtin_bit_cast(__bf16, sc[ln * SPAD + kb + 8 + ka]);
        }
        #pragma unroll
        for (int e = 0; e < 16; ++e) {
            const int kbv = half * 16 + e;      // B-layout K index
            bfr[e] = (__bf16)vb[kbv * VPAD + d];
        }
        oacc = __builtin_amdgcn_wmma_f32_16x16x32_bf16(
            false, af, false, bfr, (short)0, oacc, false, false);
#if defined(HAVE_ASYNC_LDS)
        __builtin_amdgcn_s_wait_asynccnt(0);    // staged tile written before barrier
#endif
        __syncthreads();
    }

    #pragma unroll
    for (int r = 0; r < 8; ++r) {
        const int m = r + 8 * half;
        outp[(size_t)m * DIM + d] = oacc[r];
    }
}

extern "C" void kernel_launch(void* const* d_in, const int* in_sizes, int n_in,
                              void* d_out, int out_size, void* d_ws, size_t ws_size,
                              hipStream_t stream) {
    const float* Q = (const float*)d_in[0];
    const float* K = (const float*)d_in[1];
    const float* V = (const float*)d_in[2];
    float* out = (float*)d_out;
    float* wts = out + (size_t)BATCH * SEQ * DIM;   // tuple order: (out, weights)
    dim3 grid(SEQ / 16, BATCH);
    attn_kernel<<<grid, 256, 0, stream>>>(Q, K, V, out, wts);
}
